// GraphBlock_87694642250356
// MI455X (gfx1250) — compile-verified, matrix operand order checked by reading, not run on hardware
//
#include <hip/hip_runtime.h>
#include <hip/hip_bf16.h>
#include <stdint.h>

// ---------------------------------------------------------------------------
// Scene-graph conv block for MI455X (gfx1250, wave32, WMMA).
// Shapes: DIN=128, H=256, DOUT=512, O=100000 objects, T=200000 edges.
// Fused GEMM1/2/3 per 32-edge tile (intermediates live in LDS, never HBM),
// bf16 WMMA 16x16x32 with fp32 accumulation, weights pre-packed into
// WMMA-B-fragment-native tiles, f32 atomic scatter for pooling.
// M-tile = 32: each wave holds two accumulators and reuses every B fragment
// twice. Concat buffer is split into two row-half arrays so every DS access
// stays within the 16-bit offset immediate (no VGPR address materialization).
// ---------------------------------------------------------------------------

typedef unsigned short u16;
typedef __attribute__((ext_vector_type(16))) __bf16 v16bf;
typedef __attribute__((ext_vector_type(8)))  float  v8f;

#define DIN   128
#define H     256
#define DOUT  512
#define K1    384   // 3*DIN
#define N1    768   // 3*H
#define K2    768
#define N2    512
#define K3    1024  // 2*H + DOUT
#define N3    512
#define K4    256
#define N4    512

// LDS row strides (bf16 elements), padded +8 to spread banks, keep 16B align.
#define LDA   (K1 + 8)   // 392
#define LDHP  (H  + 8)   // 264
#define LDCAT (K3 + 8)   // 1032
#define LDA4  (K4 + 8)   // 264

__device__ __forceinline__ uint32_t f2bf_bits(float x) {
    union { float f; uint32_t u; } v; v.f = x;
    return v.u + 0x7FFFu + ((v.u >> 16) & 1u);     // RNE, result in high 16
}
__device__ __forceinline__ u16 f2bf(float x) {
    return (u16)(f2bf_bits(x) >> 16);
}
__device__ __forceinline__ uint32_t pack2bf(float lo, float hi) {
    return (f2bf_bits(lo) >> 16) | (f2bf_bits(hi) & 0xFFFF0000u);
}

// A-fragment (16x32 bf16) from LDS. p = &row[k0] already biased by khalf*8.
// ISA layout: lanes<16 hold K {0..7,16..23}, lanes>=16 hold K {8..15,24..31}.
__device__ __forceinline__ v16bf load_frag_lds(const u16* p) {
    struct { uint4 a, b; } t;
    t.a = *(const uint4*)(p);
    t.b = *(const uint4*)(p + 16);
    v16bf v; __builtin_memcpy(&v, &t, 32);
    return v;
}

// Pre-packed 32x16 bf16 B-fragment: 32 contiguous bytes per lane.
__device__ __forceinline__ v16bf load_frag_packed(const u16* p) {
    struct { uint4 a, b; } t;
    t.a = *(const uint4*)(p);
    t.b = *(const uint4*)(p + 8);
    v16bf v; __builtin_memcpy(&v, &t, 32);
    return v;
}

__device__ __forceinline__ v8f wmma_bf16(v16bf a, v16bf b, v8f c) {
    return __builtin_amdgcn_wmma_f32_16x16x32_bf16(
        false, a, false, b, (short)0, c, false, false);
}

__device__ __forceinline__ void atomAddF(float* p, float v) {
#if defined(__HIP_DEVICE_COMPILE__)
    unsafeAtomicAdd(p, v);   // -> global_atomic_add_f32 (no CAS loop)
#else
    (void)p; (void)v;
#endif
}

// ---------------------------------------------------------------------------
// Weight packer: fp32 row-major W[K][N] -> bf16 WMMA-B tiles.
// Packed index: (((kt*NT + nt)*32 + lane)*16 + j); lane's 16 elements cover
// K = kt*32 + (j>>3)*16 + (lane>>4)*8 + (j&7), N = nt*16 + (lane&15).
// ---------------------------------------------------------------------------
__global__ void pack_weights_kernel(const float* __restrict__ W, int K, int N,
                                    u16* __restrict__ dst) {
    int NT = N >> 4;
    int total = K * N;
    for (int idx = blockIdx.x * blockDim.x + threadIdx.x; idx < total;
         idx += gridDim.x * blockDim.x) {
        int j    = idx & 15;
        int lane = (idx >> 4) & 31;
        int t    = idx >> 9;
        int nt   = t % NT;
        int kt   = t / NT;
        int n    = nt * 16 + (lane & 15);
        int k    = kt * 32 + ((j >> 3) << 4) + ((lane >> 4) << 3) + (j & 7);
        dst[idx] = f2bf(W[(size_t)k * N + n]);
    }
}

__global__ void zero_kernel(float4* __restrict__ p, size_t n4) {
    for (size_t i = blockIdx.x * (size_t)blockDim.x + threadIdx.x; i < n4;
         i += (size_t)gridDim.x * blockDim.x)
        p[i] = float4{0.f, 0.f, 0.f, 0.f};
}

// ---------------------------------------------------------------------------
// Edge kernel: one 256-thread block (8 wave32) per 32 edges.
// ---------------------------------------------------------------------------
__global__ __launch_bounds__(256) void edge_kernel(
    const float* __restrict__ obj, const float* __restrict__ pred,
    const int*  __restrict__ edges,
    const u16* __restrict__ pWin,  const float* __restrict__ bIn,
    const u16* __restrict__ pWedge,const float* __restrict__ bEdge,
    const u16* __restrict__ pWnode,const float* __restrict__ bNode,
    float* __restrict__ outPred, float* __restrict__ pooled,
    float* __restrict__ counts, int T)
{
    // Concat [h_s | new_pred | h_o] split by row halves: each 33KB so all
    // DS offsets fit the 16-bit immediate field.
    __shared__ __align__(16) u16 sCat0[16 * LDCAT];  // rows  0..15
    __shared__ __align__(16) u16 sCat1[16 * LDCAT];  // rows 16..31
    __shared__ __align__(16) u16 sA[32 * LDA];       // input tile (32x384) bf16
    __shared__ __align__(16) u16 sHP[32 * LDHP];     // h_p        (32x256) bf16
    __shared__ int sS[32], sOb[32];

    const int tid  = threadIdx.x;
    const int lane = tid & 31;
    const int wave = tid >> 5;
    const int col  = lane & 15;
    const int khalf= lane >> 4;
    const int rowA = lane & 15;
    const int rbase= khalf * 8;
    const int e0   = blockIdx.x * 32;

    if (tid < 32) {
        int e = e0 + tid;
        int s = 0, o = 0;
        if (e < T) {
            s = edges[2 * e];
            o = edges[2 * e + 1];
            atomAddF(&counts[s], 1.0f);
            atomAddF(&counts[o], 1.0f);
        }
        sS[tid] = s; sOb[tid] = o;
    }
    __syncthreads();

    // Gather [obj[s] | pred | obj[o]] -> bf16 LDS tile, float4-wide.
    for (int idx = tid; idx < 32 * (K1 / 4); idx += 256) {
        int r  = idx / (K1 / 4);
        int q  = idx - r * (K1 / 4);          // float4 index within row, 0..95
        int c4 = q * 4;
        int e  = e0 + r;
        float4 v = float4{0.f, 0.f, 0.f, 0.f};
        if (e < T) {
            if (q < DIN / 4)
                v = *(const float4*)&obj[(size_t)sS[r] * DIN + c4];
            else if (q < 2 * (DIN / 4))
                v = *(const float4*)&pred[(size_t)e * DIN + (c4 - DIN)];
            else
                v = *(const float4*)&obj[(size_t)sOb[r] * DIN + (c4 - 2 * DIN)];
        }
        uint2 pk;
        pk.x = pack2bf(v.x, v.y);
        pk.y = pack2bf(v.z, v.w);
        *(uint2*)&sA[r * LDA + c4] = pk;
    }
    __syncthreads();

    // ---- GEMM1: hidden = relu(A @ W_in + b_in), 48 N-tiles, K=384 ----------
    for (int t = 0; t < 6; ++t) {
        const int nt = __builtin_amdgcn_readfirstlane(wave * 6 + t);
        float bias = bIn[nt * 16 + col];
        v8f acc0, acc1;
        for (int i = 0; i < 8; ++i) { acc0[i] = bias; acc1[i] = bias; }
        const u16* ap0 = &sA[rowA * LDA] + khalf * 8;
        const u16* ap1 = ap0 + 16 * LDA;
        const u16* bp  = pWin + ((((size_t)nt) * 32 + lane) << 4);
        #pragma unroll 4
        for (int kt = 0; kt < K1 / 32; ++kt) {
            v16bf a0 = load_frag_lds(ap0 + kt * 32);
            v16bf a1 = load_frag_lds(ap1 + kt * 32);
            v16bf b  = load_frag_packed(bp);
            bp += (size_t)(N1 / 16) * 512;          // next kt
            acc0 = wmma_bf16(a0, b, acc0);
            acc1 = wmma_bf16(a1, b, acc1);
        }
        // Scalar routing: nt<16 -> h_s (cat), nt<32 -> h_p, else h_o (cat).
        u16 *dst0, *dst1; int ld;
        if (nt < 16) {
            dst0 = &sCat0[nt * 16 + col]; dst1 = &sCat1[nt * 16 + col]; ld = LDCAT;
        } else if (nt < 32) {
            dst0 = &sHP[(nt - 16) * 16 + col]; dst1 = dst0 + 16 * LDHP; ld = LDHP;
        } else {
            dst0 = &sCat0[nt * 16 + col + H]; dst1 = &sCat1[nt * 16 + col + H]; ld = LDCAT;
        }
        #pragma unroll
        for (int i = 0; i < 8; ++i) {
            dst0[(rbase + i) * ld] = f2bf(fmaxf(acc0[i], 0.0f));
            dst1[(rbase + i) * ld] = f2bf(fmaxf(acc1[i], 0.0f));
        }
    }
    __syncthreads();

    // ---- GEMM2: new_pred = relu(hidden @ W_edge + b_edge), K=768 -----------
    for (int t = 0; t < 4; ++t) {
        const int nt = __builtin_amdgcn_readfirstlane(wave * 4 + t);
        float bias = bEdge[nt * 16 + col];
        v8f acc0, acc1;
        for (int i = 0; i < 8; ++i) { acc0[i] = bias; acc1[i] = bias; }
        const u16* bp = pWedge + ((((size_t)nt) * 32 + lane) << 4);
        #pragma unroll 4
        for (int kt = 0; kt < K2 / 32; ++kt) {
            int k0 = kt * 32;
            const u16 *ap0, *ap1;
            if (k0 < H) {
                ap0 = &sCat0[rowA * LDCAT + k0];
                ap1 = &sCat1[rowA * LDCAT + k0];
            } else if (k0 < 2 * H) {
                ap0 = &sHP[rowA * LDHP + (k0 - H)];
                ap1 = ap0 + 16 * LDHP;
            } else {
                ap0 = &sCat0[rowA * LDCAT + (k0 + H)];
                ap1 = &sCat1[rowA * LDCAT + (k0 + H)];
            }
            v16bf a0 = load_frag_lds(ap0 + khalf * 8);
            v16bf a1 = load_frag_lds(ap1 + khalf * 8);
            v16bf b  = load_frag_packed(bp);
            bp += (size_t)(N2 / 16) * 512;
            acc0 = wmma_bf16(a0, b, acc0);
            acc1 = wmma_bf16(a1, b, acc1);
        }
        u16* dstc0 = &sCat0[H + nt * 16 + col];   // middle of concat
        u16* dstc1 = &sCat1[H + nt * 16 + col];
        #pragma unroll
        for (int i = 0; i < 8; ++i) {
            int row = rbase + i;
            float v0 = fmaxf(acc0[i], 0.0f);
            float v1 = fmaxf(acc1[i], 0.0f);
            if (e0 + row < T)      outPred[(size_t)(e0 + row)      * DOUT + nt * 16 + col] = v0;
            if (e0 + row + 16 < T) outPred[(size_t)(e0 + row + 16) * DOUT + nt * 16 + col] = v1;
            dstc0[row * LDCAT] = f2bf(v0);
            dstc1[row * LDCAT] = f2bf(v1);
        }
    }
    __syncthreads();

    // ---- GEMM3: new_t = relu(concat @ W_node + b_node), K=1024, scatter ----
    for (int t = 0; t < 4; ++t) {
        const int nt = __builtin_amdgcn_readfirstlane(wave * 4 + t);
        float bias = bNode[nt * 16 + col];
        v8f acc0, acc1;
        for (int i = 0; i < 8; ++i) { acc0[i] = bias; acc1[i] = bias; }
        const u16* ap0 = &sCat0[rowA * LDCAT] + khalf * 8;
        const u16* ap1 = &sCat1[rowA * LDCAT] + khalf * 8;
        const u16* bp  = pWnode + ((((size_t)nt) * 32 + lane) << 4);
        #pragma unroll 4
        for (int kt = 0; kt < K3 / 32; ++kt) {
            v16bf a0 = load_frag_lds(ap0 + kt * 32);
            v16bf a1 = load_frag_lds(ap1 + kt * 32);
            v16bf b  = load_frag_packed(bp);
            bp += (size_t)(N3 / 16) * 512;
            acc0 = wmma_bf16(a0, b, acc0);
            acc1 = wmma_bf16(a1, b, acc1);
        }
        // Scalar routing: nt<16 -> new_s via s_idx, else new_o via o_idx.
        const int* idxArr = (nt < 16) ? sS : sOb;
        const int cofs = ((nt & 15) * 16) + col;   // column within H
        #pragma unroll
        for (int i = 0; i < 8; ++i) {
            int row0 = rbase + i;
            int row1 = row0 + 16;
            if (e0 + row0 < T)
                atomAddF(&pooled[(size_t)idxArr[row0] * H + cofs],
                         fmaxf(acc0[i], 0.0f));
            if (e0 + row1 < T)
                atomAddF(&pooled[(size_t)idxArr[row1] * H + cofs],
                         fmaxf(acc1[i], 0.0f));
        }
    }
}

// ---------------------------------------------------------------------------
// Node kernel: new_obj = relu((pooled / clip(counts,1)) @ W_out + b_out)
// One 256-thread block per 32 objects; two accumulators per wave.
// ---------------------------------------------------------------------------
__global__ __launch_bounds__(256) void node_kernel(
    const float* __restrict__ pooled, const float* __restrict__ counts,
    const u16* __restrict__ pWout, const float* __restrict__ bOut,
    float* __restrict__ outObj, int O)
{
    __shared__ __align__(16) u16 sA[32 * LDA4];
    __shared__ float sInv[32];

    const int tid  = threadIdx.x;
    const int lane = tid & 31;
    const int wave = tid >> 5;
    const int col  = lane & 15;
    const int khalf= lane >> 4;
    const int rowA = lane & 15;
    const int rbase= khalf * 8;
    const int o0   = blockIdx.x * 32;

    if (tid < 32) {
        int ob = o0 + tid;
        float c = (ob < O) ? counts[ob] : 1.0f;
        sInv[tid] = 1.0f / fmaxf(c, 1.0f);
    }
    __syncthreads();

    for (int idx = tid; idx < 32 * (K4 / 4); idx += 256) {
        int r  = idx >> 6;
        int c4 = (idx & 63) * 4;
        int ob = o0 + r;
        float4 v = float4{0.f, 0.f, 0.f, 0.f};
        if (ob < O) v = *(const float4*)&pooled[(size_t)ob * H + c4];
        float s = sInv[r];
        uint2 pk;
        pk.x = pack2bf(v.x * s, v.y * s);
        pk.y = pack2bf(v.z * s, v.w * s);
        *(uint2*)&sA[r * LDA4 + c4] = pk;
    }
    __syncthreads();

    for (int t = 0; t < 4; ++t) {
        const int nt = __builtin_amdgcn_readfirstlane(wave * 4 + t);
        float bias = bOut[nt * 16 + col];
        v8f acc0, acc1;
        for (int i = 0; i < 8; ++i) { acc0[i] = bias; acc1[i] = bias; }
        const u16* ap0 = &sA[rowA * LDA4] + khalf * 8;
        const u16* ap1 = ap0 + 16 * LDA4;
        const u16* bp  = pWout + ((((size_t)nt) * 32 + lane) << 4);
        #pragma unroll
        for (int kt = 0; kt < K4 / 32; ++kt) {
            v16bf a0 = load_frag_lds(ap0 + kt * 32);
            v16bf a1 = load_frag_lds(ap1 + kt * 32);
            v16bf b  = load_frag_packed(bp);
            bp += (size_t)(N4 / 16) * 512;
            acc0 = wmma_bf16(a0, b, acc0);
            acc1 = wmma_bf16(a1, b, acc1);
        }
        #pragma unroll
        for (int i = 0; i < 8; ++i) {
            int row0 = rbase + i;
            int row1 = row0 + 16;
            if (o0 + row0 < O)
                outObj[(size_t)(o0 + row0) * DOUT + nt * 16 + col] = fmaxf(acc0[i], 0.0f);
            if (o0 + row1 < O)
                outObj[(size_t)(o0 + row1) * DOUT + nt * 16 + col] = fmaxf(acc1[i], 0.0f);
        }
    }
}

// ---------------------------------------------------------------------------
extern "C" void kernel_launch(void* const* d_in, const int* in_sizes, int n_in,
                              void* d_out, int out_size, void* d_ws, size_t ws_size,
                              hipStream_t stream) {
    const float* obj    = (const float*)d_in[0];
    const float* pred   = (const float*)d_in[1];
    const int*   edges  = (const int*)  d_in[2];
    const float* W_in   = (const float*)d_in[3];
    const float* b_in   = (const float*)d_in[4];
    const float* W_edge = (const float*)d_in[5];
    const float* b_edge = (const float*)d_in[6];
    const float* W_node = (const float*)d_in[7];
    const float* b_node = (const float*)d_in[8];
    const float* W_out  = (const float*)d_in[9];
    const float* b_out  = (const float*)d_in[10];

    const int O = in_sizes[0] / DIN;
    const int T = in_sizes[1] / DIN;

    // Workspace layout (all 16B aligned)
    char* ws = (char*)d_ws;
    size_t off = 0;
    u16* pWin   = (u16*)(ws + off); off += (size_t)K1 * N1 * sizeof(u16);
    u16* pWedge = (u16*)(ws + off); off += (size_t)K2 * N2 * sizeof(u16);
    u16* pWnode = (u16*)(ws + off); off += (size_t)K3 * N3 * sizeof(u16);
    u16* pWout  = (u16*)(ws + off); off += (size_t)K4 * N4 * sizeof(u16);
    float* pooled = (float*)(ws + off); off += (size_t)O * H * sizeof(float);
    float* counts = (float*)(ws + off); off += (size_t)O * sizeof(float);

    float* outObj  = (float*)d_out;                    // (O, 512)
    float* outPred = outObj + (size_t)O * DOUT;        // (T, 512)

    // 1) pack weights to bf16 WMMA-B tiles
    pack_weights_kernel<<<(K1*N1 + 255)/256, 256, 0, stream>>>(W_in,   K1, N1, pWin);
    pack_weights_kernel<<<(K2*N2 + 255)/256, 256, 0, stream>>>(W_edge, K2, N2, pWedge);
    pack_weights_kernel<<<(K3*N3 + 255)/256, 256, 0, stream>>>(W_node, K3, N3, pWnode);
    pack_weights_kernel<<<(K4*N4 + 255)/256, 256, 0, stream>>>(W_out,  K4, N4, pWout);

    // 2) zero pooled + counts (contiguous floats; total count divisible by 4)
    {
        size_t ntot = (size_t)O * (H + 1);
        zero_kernel<<<4096, 256, 0, stream>>>((float4*)pooled, ntot / 4);
    }

    // 3) fused edge pipeline (GEMM1/2/3 + scatter)
    edge_kernel<<<(T + 31)/32, 256, 0, stream>>>(
        obj, pred, edges, pWin, b_in, pWedge, b_edge, pWnode, b_node,
        outPred, pooled, counts, T);

    // 4) node update (GEMM4)
    node_kernel<<<(O + 31)/32, 256, 0, stream>>>(
        pooled, counts, pWout, b_out, outObj, O);
}
